// StaticRecurrentEntNet_50689204027760
// MI455X (gfx1250) — matrix operand order repaired and checked by reading, not compiled, not run on hardware
//
#include <hip/hip_runtime.h>

// Problem constants (from reference)
#define B_ 256
#define S_ 32
#define L_ 32
#define D_ 256
#define E_ 64

typedef __attribute__((ext_vector_type(16))) __bf16 v16bf;
typedef __attribute__((ext_vector_type(8)))  __bf16 v8bf;
typedef __attribute__((ext_vector_type(8)))  float  v8f;

static __device__ __forceinline__ __bf16 to_bf16(float x) { return (__bf16)x; }

// ---------------------------------------------------------------------------
// Kernel 1: enc[b,s,d] = sum_l emb[idx[b,s,l], d] * mask[b,s,l]
//           active[b,s] = (sum_l mask) > 0
// Gather-reduce, memory bound (~268 MB reads). One block per (b,s), d=thread.
// ---------------------------------------------------------------------------
__global__ void encode_kernel(const int* __restrict__ idx, const float* __restrict__ mask,
                              const float* __restrict__ emb, float* __restrict__ enc,
                              int* __restrict__ active) {
  const int bs = blockIdx.x;
  const int d  = threadIdx.x;
  const int* ids = idx + bs * L_;
  const float* mp = mask + bs * L_;
  float acc = 0.f, msum = 0.f;
#pragma unroll
  for (int l = 0; l < L_; ++l) {
    const float mv = mp[l];
    msum += mv;
    acc += emb[(long)ids[l] * D_ + d] * mv;
  }
  enc[(long)bs * D_ + d] = acc;
  if (d == 0) active[bs] = (msum > 0.f) ? 1 : 0;
}

// ---------------------------------------------------------------------------
// Kernel 2: transpose + convert U,V,W (f32, KxN row-major) -> Xt (bf16, NxK)
// so B-fragments become contiguous 32B loads along K.
// ---------------------------------------------------------------------------
__global__ void tconv_kernel(const float* __restrict__ U, const float* __restrict__ V,
                             const float* __restrict__ W, __bf16* __restrict__ Ut,
                             __bf16* __restrict__ Vt, __bf16* __restrict__ Wt) {
  const int mat = blockIdx.x / D_;
  const int n   = blockIdx.x % D_;
  const int k   = threadIdx.x;
  const float* src = (mat == 0) ? U : (mat == 1) ? V : W;
  __bf16*      dst = (mat == 0) ? Ut : (mat == 1) ? Vt : Wt;
  dst[n * D_ + k] = to_bf16(src[k * D_ + n]);
}

// ---------------------------------------------------------------------------
// Kernel 3: C[M,256] = A[M,256] (f32 -> bf16) x B (given as Bt[N,K] bf16)
// Block = 256 threads = 8 wave32, computes a 64-row slab.
// Tiling: wave w owns M-tile (w>>1), N-tiles ((w&1)*8 .. +7). 16x16x32 WMMA.
// Used for kV = keys @ V (M=16384) and eW = enc @ W (M=8192), L2-resident B.
// ---------------------------------------------------------------------------
__global__ void gemm_bf16_kernel(const float* __restrict__ A, const __bf16* __restrict__ Bt,
                                 float* __restrict__ C) {
  __shared__ __bf16 As[64 * D_];
  const int tid = threadIdx.x;
  const int m0  = blockIdx.x * 64;
  for (int i = tid; i < 64 * D_; i += 256)
    As[i] = to_bf16(A[(long)m0 * D_ + i]);
  __syncthreads();

  const int wave = tid >> 5;      // wave32
  const int lane = tid & 31;
  const int m  = lane & 15;       // A: row-in-tile / B,C: col-in-tile
  const int hi = lane >> 4;       // lane half
  const int mt = wave >> 1;

  for (int j = 0; j < 8; ++j) {
    const int nt  = ((wave & 1) << 3) + j;
    const int col = nt * 16 + m;
    v8f c = {0.f, 0.f, 0.f, 0.f, 0.f, 0.f, 0.f, 0.f};
#pragma unroll
    for (int kk = 0; kk < 8; ++kk) {
      // A frag: lane half hi -> K offsets {hi*16..+15} within 32-wide K tile
      const __bf16* ap = &As[(mt * 16 + m) * D_ + kk * 32 + hi * 16];
      v8bf alo = *(const v8bf*)ap;
      v8bf ahi = *(const v8bf*)(ap + 8);
      v16bf a = __builtin_shufflevector(alo, ahi, 0,1,2,3,4,5,6,7,8,9,10,11,12,13,14,15);
      // B frag: col fixed per lane, 16 contiguous K values (32B load from Bt)
      v16bf bfr = *(const v16bf*)&Bt[(long)col * D_ + kk * 32 + hi * 16];
      c = __builtin_amdgcn_wmma_f32_16x16x32_bf16(false, a, false, bfr, (short)0, c, false, false);
    }
#pragma unroll
    for (int i = 0; i < 8; ++i) {
      const int row = m0 + mt * 16 + hi * 8 + i;   // C layout: v[i] -> M = hi*8+i
      C[(long)row * D_ + col] = c[i];
    }
  }
}

// ---------------------------------------------------------------------------
// Kernel 4: the 32-step recurrence. One workgroup (256 thr = 8 waves) per batch.
// h (64x256 f32) lives in LDS the whole scan; Ut cached in LDS (bf16).
// Per step: gate g -> h_bf16 snapshot -> WMMA h@U (+kV+eW as C-init) ->
// relu/gate-blend/L2-normalize, all in LDS. ~236 KB dynamic LDS.
// ---------------------------------------------------------------------------
__global__ void scan_kernel(const float* __restrict__ keys, const __bf16* __restrict__ Ut,
                            const float* __restrict__ kV, const float* __restrict__ enc,
                            const float* __restrict__ eW, const int* __restrict__ active,
                            float* __restrict__ hout) {
  extern __shared__ char smem[];
  float*  h     = (float*)(smem);                 // 65536 B : hidden state f32
  __bf16* hb    = (__bf16*)(smem + 65536);        // 32768 B : bf16 snapshot (WMMA A)
  __bf16* UtS   = (__bf16*)(smem + 98304);        // 131072 B: U transposed bf16 (WMMA B)
  float*  encS  = (float*)(smem + 229376);        // 1024 B
  float*  eWs   = (float*)(smem + 230400);        // 1024 B
  float*  gpart = (float*)(smem + 231424);        // 1024 B : gate partials [64][4]
  float*  gv    = (float*)(smem + 232448);        // 256 B  : gate, then inv-norm
  float*  sqp   = (float*)(smem + 232704);        // 8192 B : sq-sum partials [64][32]
  // total = 240896 bytes

  const int b    = blockIdx.x;
  const int tid  = threadIdx.x;
  const int wave = tid >> 5, lane = tid & 31;
  const int m = lane & 15, hi = lane >> 4, mt = wave >> 1;

  for (int i = tid; i < E_ * D_; i += 256) h[i] = 0.f;
  for (int i = tid; i < D_ * D_; i += 256) UtS[i] = Ut[i];
  __syncthreads();

  const float* keyb = keys + (long)b * E_ * D_;
  const float* kVb  = kV   + (long)b * E_ * D_;

  for (int s = 0; s < S_; ++s) {
    if (active[b * S_ + s]) {   // block-uniform branch
      // stage per-step vectors
      encS[tid] = enc[((long)b * S_ + s) * D_ + tid];
      eWs[tid]  = eW[((long)b * S_ + s) * D_ + tid];
      __syncthreads();
      // gate partials: g[e] = sigmoid(sum_d enc_d*(h[e,d]+key[e,d]))
      {
        const int e = tid >> 2, q = tid & 3;
        const float* hr = h + e * D_ + q * 64;
        const float* kr = keyb + e * D_ + q * 64;
        const float* er = encS + q * 64;
        float p = 0.f;
#pragma unroll 8
        for (int d = 0; d < 64; ++d) p += er[d] * (hr[d] + kr[d]);
        gpart[tid] = p;
      }
      __syncthreads();
      if (tid < E_) {
        const float t = gpart[tid*4] + gpart[tid*4+1] + gpart[tid*4+2] + gpart[tid*4+3];
        gv[tid] = 1.f / (1.f + __expf(-t));
      }
      for (int i = tid; i < E_ * D_; i += 256) hb[i] = to_bf16(h[i]);
      __syncthreads();

      // h_t = relu(h@U + kV + eW); h := h + g*h_t (unnormalized); sq partials
      float rowsq[8];
#pragma unroll
      for (int i = 0; i < 8; ++i) rowsq[i] = 0.f;
      for (int j = 0; j < 8; ++j) {
        const int nt  = ((wave & 1) << 3) + j;
        const int col = nt * 16 + m;
        v8f c;
#pragma unroll
        for (int i = 0; i < 8; ++i) {
          const int row = mt * 16 + hi * 8 + i;
          c[i] = kVb[row * D_ + col] + eWs[col];   // C-init = invariant terms
        }
#pragma unroll
        for (int kk = 0; kk < 8; ++kk) {
          const __bf16* ap = &hb[(mt * 16 + m) * D_ + kk * 32 + hi * 16];
          v8bf alo = *(const v8bf*)ap;
          v8bf ahi = *(const v8bf*)(ap + 8);
          v16bf a = __builtin_shufflevector(alo, ahi, 0,1,2,3,4,5,6,7,8,9,10,11,12,13,14,15);
          v16bf bfr = *(const v16bf*)&UtS[(nt * 16 + m) * D_ + kk * 32 + hi * 16];
          c = __builtin_amdgcn_wmma_f32_16x16x32_bf16(false, a, false, bfr, (short)0, c, false, false);
        }
#pragma unroll
        for (int i = 0; i < 8; ++i) {
          const int row = mt * 16 + hi * 8 + i;
          const float ht = c[i] > 0.f ? c[i] : 0.f;
          const float hn = h[row * D_ + col] + gv[row] * ht;  // each elem owned by one lane
          h[row * D_ + col] = hn;
          rowsq[i] += hn * hn;
        }
      }
      {
        const int contrib = ((wave & 1) << 4) + m;  // 0..31, unique per (row, contrib)
#pragma unroll
        for (int i = 0; i < 8; ++i)
          sqp[(mt * 16 + hi * 8 + i) * 32 + contrib] = rowsq[i];
      }
      __syncthreads();
      if (tid < E_) {
        float ss = 0.f;
#pragma unroll 8
        for (int k = 0; k < 32; ++k) ss += sqp[tid * 32 + k];
        gv[tid] = rsqrtf(fmaxf(ss, 1e-12f));        // reuse gv as inv-norm
      }
      __syncthreads();
      for (int i = tid; i < E_ * D_; i += 256) h[i] *= gv[i >> 8];
      __syncthreads();
    }
  }
  for (int i = tid; i < E_ * D_; i += 256) hout[(long)b * E_ * D_ + i] = h[i];
}

// ---------------------------------------------------------------------------
// Host launcher. Workspace need: enc 8MB + act 32KB + 3x128KB bf16 weights
// + kV 16.8MB + eW 8MB ~= 34 MB.
// ---------------------------------------------------------------------------
extern "C" void kernel_launch(void* const* d_in, const int* in_sizes, int n_in,
                              void* d_out, int out_size, void* d_ws, size_t ws_size,
                              hipStream_t stream) {
  (void)in_sizes; (void)n_in; (void)out_size; (void)ws_size;
  const int*   idx  = (const int*)d_in[0];
  const float* mask = (const float*)d_in[1];
  const float* keys = (const float*)d_in[2];
  const float* emb  = (const float*)d_in[3];
  const float* U    = (const float*)d_in[4];
  const float* V    = (const float*)d_in[5];
  const float* W    = (const float*)d_in[6];
  float* out = (float*)d_out;

  char* ws = (char*)d_ws;
  size_t off = 0;
  auto alloc = [&](size_t bytes) -> char* {
    char* p = ws + off;
    off += (bytes + 255) & ~(size_t)255;
    return p;
  };
  float*  enc = (float*) alloc((size_t)B_ * S_ * D_ * 4);
  int*    act = (int*)   alloc((size_t)B_ * S_ * 4);
  __bf16* Ut  = (__bf16*)alloc((size_t)D_ * D_ * 2);
  __bf16* Vt  = (__bf16*)alloc((size_t)D_ * D_ * 2);
  __bf16* Wt  = (__bf16*)alloc((size_t)D_ * D_ * 2);
  float*  kV  = (float*) alloc((size_t)B_ * E_ * D_ * 4);
  float*  eWs = (float*) alloc((size_t)B_ * S_ * D_ * 4);

  tconv_kernel<<<3 * D_, D_, 0, stream>>>(U, V, W, Ut, Vt, Wt);
  encode_kernel<<<B_ * S_, D_, 0, stream>>>(idx, mask, emb, enc, act);
  gemm_bf16_kernel<<<(B_ * E_) / 64, 256, 0, stream>>>(keys, Vt, kV);   // kV = keys @ V
  gemm_bf16_kernel<<<(B_ * S_) / 64, 256, 0, stream>>>(enc, Wt, eWs);   // eW = enc @ W
  scan_kernel<<<B_, 256, 240896, stream>>>(keys, Ut, kV, enc, eWs, act, out);
}
